// GIN_27977416966468
// MI455X (gfx1250) — matrix operand order repaired and checked by reading, not compiled, not run on hardware
//
#include <hip/hip_runtime.h>

// Problem constants (fixed by the reference)
#define NN    100000   // nodes
#define EE    3200000  // edges
#define FF    128      // input features
#define HH    256      // hidden dim
#define DD    256      // embedding dim
#define BN_EPS 1e-5f

// ---- CDNA5 WMMA types (wave32; 16x16x32 bf16 -> f32) ----
typedef __attribute__((ext_vector_type(16))) __bf16 v16bf;
typedef __attribute__((ext_vector_type(8)))  float  v8f;

union FragU { uint4 u[2]; v16bf v; };

__device__ __forceinline__ unsigned short f32_to_bf16_rne(float f) {
    union { float f; unsigned int u; } c; c.f = f;
    unsigned int u = c.u;
    u += 0x7FFFu + ((u >> 16) & 1u);   // round-to-nearest-even
    return (unsigned short)(u >> 16);
}

// ---------------- prep kernels (tiny, once per launch) ----------------

// W1t[j*fin+i] = bf16( W1[i*fout+j] * g[j]*rsqrt(var[j]+eps) )   (BN scale folded, transposed)
__global__ void k_prep_w1t(const float* __restrict__ W1, const float* __restrict__ g,
                           const float* __restrict__ var, unsigned short* __restrict__ W1t,
                           int fin, int fout) {
    int idx = blockIdx.x * blockDim.x + threadIdx.x;
    if (idx >= fin * fout) return;
    int j = idx % fout;
    int i = idx / fout;
    float s = g[j] * rsqrtf(var[j] + BN_EPS);
    W1t[(long long)j * fin + i] = f32_to_bf16_rne(W1[(long long)i * fout + j] * s);
}

// b1f[j] = (b1[j]-mu[j]) * g[j]*rsqrt(var[j]+eps) + be[j]   (BN shift folded)
__global__ void k_prep_b1(const float* __restrict__ b1, const float* __restrict__ g,
                          const float* __restrict__ be, const float* __restrict__ mu,
                          const float* __restrict__ var, float* __restrict__ b1f, int fout) {
    int j = blockIdx.x * blockDim.x + threadIdx.x;
    if (j >= fout) return;
    float s = g[j] * rsqrtf(var[j] + BN_EPS);
    b1f[j] = (b1[j] - mu[j]) * s + be[j];
}

// W2t[j*fin+i] = bf16( W2[i*fout+j] )   (transpose + convert)
__global__ void k_prep_w2t(const float* __restrict__ W2, unsigned short* __restrict__ W2t,
                           int fin, int fout) {
    int idx = blockIdx.x * blockDim.x + threadIdx.x;
    if (idx >= fin * fout) return;
    int j = idx % fout;
    int i = idx / fout;
    W2t[(long long)j * fin + i] = f32_to_bf16_rne(W2[(long long)i * fout + j]);
}

// ---------------- aggregation: z = h + segment_sum(h[src] -> dst) ----------------

__global__ void k_copy_f32v4(const float4* __restrict__ src, float4* __restrict__ dst, int n4) {
    int i = blockIdx.x * blockDim.x + threadIdx.x;
    if (i < n4) dst[i] = src[i];
}

// One thread per (edge, 4-feature chunk): gather float4 from h[src], atomic-add into z[dst].
// z is L2-resident (<= 102 MB vs 192 MB L2) so atomics stay on-die.
__global__ void k_edge_agg(const float* __restrict__ h, const int* __restrict__ src,
                           const int* __restrict__ dst, float* __restrict__ z, int F) {
    long long idx = (long long)blockIdx.x * blockDim.x + threadIdx.x;
    int chunks = F >> 2;
    long long total = (long long)EE * chunks;
    if (idx >= total) return;
    int e = (int)(idx / chunks);
    int c = ((int)(idx % chunks)) << 2;
    int s = src[e], d = dst[e];
    const float4 hv = *(const float4*)(h + (long long)s * F + c);
    float* zp = z + (long long)d * F + c;
    atomicAdd(zp + 0, hv.x);
    atomicAdd(zp + 1, hv.y);
    atomicAdd(zp + 2, hv.z);
    atomicAdd(zp + 3, hv.w);
}

__global__ void k_f32_to_bf16(const float* __restrict__ src, unsigned short* __restrict__ dst, int n) {
    int i = blockIdx.x * blockDim.x + threadIdx.x;
    if (i < n) dst[i] = f32_to_bf16_rne(src[i]);
}

// ---------------- WMMA GEMM: C = act(A(bf16, MxK) @ Bt^T(bf16, NnxK) + bias) ----------------
// 256 threads = 8 waves; wave w -> rows [m0, m0+16), 4 column tiles of 16 -> 16x64 strip.
// A fragment (contiguous k = k0 + 16*half + s, same order as B): lane r(0..15) row m0+r, half=lane>>4:
//   elems 0..7  = A[row][k0 + half*16 .. +7]       (VGPR0..3)
//   elems 8..15 = A[row][k0 + half*16 + 8 .. +7]   (VGPR4..7)
// B fragment (16-bit B 32x16): lane col n, elems 0..15 = K = k0 + half*16 .. +15 (contiguous in Bt row).
// C/D f32 16x16: VGPR v -> row m0 + v + 8*half, col = n0 + r.
__global__ __launch_bounds__(256, 2)
void k_wmma_gemm(const unsigned short* __restrict__ A,   // M x K, bf16 row-major
                 const unsigned short* __restrict__ Bt,  // Nn x K, bf16 (weights pre-transposed)
                 const float* __restrict__ bias,         // Nn
                 float* __restrict__ Cf,                 // optional f32 out (M x Nn)
                 unsigned short* __restrict__ Cb,        // optional bf16 out (M x Nn)
                 int M, int K, int Nn, int relu) {
    const int wave = threadIdx.x >> 5;
    const int lane = threadIdx.x & 31;
    const int r    = lane & 15;
    const int half = lane >> 4;
    const int m0   = blockIdx.x * 128 + wave * 16;
    const int n0   = blockIdx.y * 64;
    if (m0 + 16 > M) return;  // wave-uniform: EXEC stays all-ones for WMMA

    v8f acc0 = {}, acc1 = {}, acc2 = {}, acc3 = {};

    const unsigned short* arow = A  + (long long)(m0 + r) * K;
    const unsigned short* br0  = Bt + (long long)(n0 +      r) * K;
    const unsigned short* br1  = Bt + (long long)(n0 + 16 + r) * K;
    const unsigned short* br2  = Bt + (long long)(n0 + 32 + r) * K;
    const unsigned short* br3  = Bt + (long long)(n0 + 48 + r) * K;

    for (int k0 = 0; k0 < K; k0 += 32) {
        FragU a;
        a.u[0] = *(const uint4*)(arow + k0 + half * 16);
        a.u[1] = *(const uint4*)(arow + k0 + half * 16 + 8);
        __builtin_prefetch(arow + k0 + 64, 0, 1);  // global_prefetch_b8 for next A tile

        const int kb = k0 + half * 16;
        FragU fb0, fb1, fb2, fb3;
        fb0.u[0] = *(const uint4*)(br0 + kb); fb0.u[1] = *(const uint4*)(br0 + kb + 8);
        fb1.u[0] = *(const uint4*)(br1 + kb); fb1.u[1] = *(const uint4*)(br1 + kb + 8);
        fb2.u[0] = *(const uint4*)(br2 + kb); fb2.u[1] = *(const uint4*)(br2 + kb + 8);
        fb3.u[0] = *(const uint4*)(br3 + kb); fb3.u[1] = *(const uint4*)(br3 + kb + 8);

        acc0 = __builtin_amdgcn_wmma_f32_16x16x32_bf16(false, a.v, false, fb0.v, (short)0, acc0, false, false);
        acc1 = __builtin_amdgcn_wmma_f32_16x16x32_bf16(false, a.v, false, fb1.v, (short)0, acc1, false, false);
        acc2 = __builtin_amdgcn_wmma_f32_16x16x32_bf16(false, a.v, false, fb2.v, (short)0, acc2, false, false);
        acc3 = __builtin_amdgcn_wmma_f32_16x16x32_bf16(false, a.v, false, fb3.v, (short)0, acc3, false, false);
    }

    const float bia0 = bias[n0 +      r];
    const float bia1 = bias[n0 + 16 + r];
    const float bia2 = bias[n0 + 32 + r];
    const float bia3 = bias[n0 + 48 + r];

    #pragma unroll
    for (int v = 0; v < 8; ++v) {
        const int row = m0 + v + half * 8;
        float x0 = acc0[v] + bia0;
        float x1 = acc1[v] + bia1;
        float x2 = acc2[v] + bia2;
        float x3 = acc3[v] + bia3;
        if (relu) {
            x0 = fmaxf(x0, 0.f); x1 = fmaxf(x1, 0.f);
            x2 = fmaxf(x2, 0.f); x3 = fmaxf(x3, 0.f);
        }
        const long long base = (long long)row * Nn + n0 + r;
        if (Cf) {
            Cf[base] = x0; Cf[base + 16] = x1; Cf[base + 32] = x2; Cf[base + 48] = x3;
        }
        if (Cb) {
            Cb[base]      = f32_to_bf16_rne(x0);
            Cb[base + 16] = f32_to_bf16_rne(x1);
            Cb[base + 32] = f32_to_bf16_rne(x2);
            Cb[base + 48] = f32_to_bf16_rne(x3);
        }
    }
}

// ---------------- final projection: out = h3 @ out_W(256x2) + out_b ----------------
__global__ void k_out_gemv(const float* __restrict__ h, const float* __restrict__ W,
                           const float* __restrict__ b, float* __restrict__ out) {
    int n = blockIdx.x * blockDim.x + threadIdx.x;
    if (n >= NN) return;
    const float* hr = h + (long long)n * DD;
    float o0 = 0.f, o1 = 0.f;
    #pragma unroll 8
    for (int k = 0; k < DD; ++k) {
        float hv = hr[k];
        o0 += hv * W[2 * k];
        o1 += hv * W[2 * k + 1];
    }
    out[2 * n]     = o0 + b[0];
    out[2 * n + 1] = o1 + b[1];
}

// ---------------- launch ----------------
extern "C" void kernel_launch(void* const* d_in, const int* in_sizes, int n_in,
                              void* d_out, int out_size, void* d_ws, size_t ws_size,
                              hipStream_t stream) {
    (void)in_sizes; (void)n_in; (void)out_size; (void)ws_size;

    const float* x   = (const float*)d_in[0];
    const int*   ei  = (const int*)d_in[1];
    const int*   src = ei;
    const int*   dst = ei + EE;
    const float* p[3][8];
    for (int l = 0; l < 3; ++l)
        for (int k = 0; k < 8; ++k)
            p[l][k] = (const float*)d_in[2 + l * 8 + k];
    // p[l]: 0=W1 1=b1 2=g 3=be 4=mu 5=var 6=W2 7=b2
    const float* outW = (const float*)d_in[26];
    const float* outB = (const float*)d_in[27];

    // workspace carve-up (256B aligned)
    char* ws = (char*)d_ws;
    size_t off = 0;
    auto carve = [&](size_t bytes) -> void* {
        void* ptr = ws + off;
        off = (off + bytes + 255) & ~(size_t)255;
        return ptr;
    };
    const size_t NB = (size_t)NN * 256;
    float*          bufZ  = (float*)carve(NB * 4);          // z (f32)
    float*          bufH  = (float*)carve(NB * 4);          // layer output (f32)
    unsigned short* bufZb = (unsigned short*)carve(NB * 2); // z (bf16)
    unsigned short* bufY  = (unsigned short*)carve(NB * 2); // hidden (bf16)
    unsigned short* W1t[3]; float* b1f[3]; unsigned short* W2t[3];
    const int fins[3] = { FF, HH, HH };
    for (int l = 0; l < 3; ++l) {
        W1t[l] = (unsigned short*)carve((size_t)fins[l] * HH * 2);
        b1f[l] = (float*)carve((size_t)HH * 4);
        W2t[l] = (unsigned short*)carve((size_t)HH * 256 * 2);
    }

    // prep weights (fold BN, transpose, bf16)
    for (int l = 0; l < 3; ++l) {
        int fin = fins[l];
        int ne1 = fin * HH;
        k_prep_w1t<<<(ne1 + 255) / 256, 256, 0, stream>>>(p[l][0], p[l][2], p[l][5], W1t[l], fin, HH);
        k_prep_b1<<<1, 256, 0, stream>>>(p[l][1], p[l][2], p[l][3], p[l][4], p[l][5], b1f[l], HH);
        int ne2 = HH * 256;
        k_prep_w2t<<<(ne2 + 255) / 256, 256, 0, stream>>>(p[l][6], W2t[l], HH, 256);
    }

    const dim3 gemmGrid((NN + 127) / 128, 256 / 64);
    const float* hprev = x;
    for (int l = 0; l < 3; ++l) {
        const int F = fins[l];
        // z = h_prev
        int n4 = NN * F / 4;
        k_copy_f32v4<<<(n4 + 255) / 256, 256, 0, stream>>>((const float4*)hprev, (float4*)bufZ, n4);
        // z[dst] += h_prev[src]  (scatter, L2-resident atomics)
        long long tot = (long long)EE * (F / 4);
        int blocks = (int)((tot + 255) / 256);
        k_edge_agg<<<blocks, 256, 0, stream>>>(hprev, src, dst, bufZ, F);
        // z -> bf16
        int ne = NN * F;
        k_f32_to_bf16<<<(ne + 255) / 256, 256, 0, stream>>>(bufZ, bufZb, ne);
        // GEMM1: relu(z @ W1' + b1')  -> bf16
        k_wmma_gemm<<<gemmGrid, 256, 0, stream>>>(bufZb, W1t[l], b1f[l],
                                                  nullptr, bufY, NN, F, 256, 1);
        // GEMM2: (relu_l<2)(y @ W2 + b2) -> f32
        k_wmma_gemm<<<gemmGrid, 256, 0, stream>>>(bufY, W2t[l], p[l][7],
                                                  bufH, nullptr, NN, 256, 256, (l < 2) ? 1 : 0);
        hprev = bufH;
    }

    // final linear head
    k_out_gemv<<<(NN + 255) / 256, 256, 0, stream>>>(bufH, outW, outB, (float*)d_out);
}
